// Gaussian2D_4870492914135
// MI455X (gfx1250) — matrix operand order repaired and checked by reading, not compiled, not run on hardware
//
#include <hip/hip_runtime.h>
#include <math.h>

// CDNA5 / gfx1250: wave32, WMMA 16x16x32 f16 -> f32
typedef __attribute__((ext_vector_type(16))) _Float16 v16h;
typedef __attribute__((ext_vector_type(8)))  float    v8f;

#define NG    1024
#define IMG_H 256
#define IMG_W 256
#define NCH   3                     // wgt = alpha * sigmoid(colors[:, :3]) -> 3 channels

// workspace layout:
//   [0, 8*NG) floats          : per-gaussian params AoS, 8 floats each:
//                               {cx, cy, inv00, inv01+inv10, inv11, norm, 0, 0}
//   [8*NG, 8*NG + 8192) floats: packed f16 B-operand table,
//                               32 chunks x 32 lanes x 16 halves (32 KB)
#define PARAM_FLOATS (8 * NG)

// ---------------------------------------------------------------------------
// Kernel 1: per-gaussian covariance preprocessing.
// cov = R (s s^T) R^T + I ; store inverse terms + normalization (AoS, 32B).
// ---------------------------------------------------------------------------
__global__ __launch_bounds__(256) void gs2d_prep(
    const float* __restrict__ scale,     // (N,2,1)
    const float* __restrict__ rot,       // (N,2,2)
    const float* __restrict__ means,     // (N,2,1)
    float* __restrict__ ws)
{
    int g = blockIdx.x * blockDim.x + threadIdx.x;
    if (g >= NG) return;

    float s0 = scale[g * 2 + 0], s1 = scale[g * 2 + 1];
    float r00 = rot[g * 4 + 0], r01 = rot[g * 4 + 1];
    float r10 = rot[g * 4 + 2], r11 = rot[g * 4 + 3];

    float ss00 = s0 * s0, ss01 = s0 * s1, ss11 = s1 * s1;
    // T = ssT * R^T
    float T00 = ss00 * r00 + ss01 * r01;
    float T01 = ss00 * r10 + ss01 * r11;
    float T10 = ss01 * r00 + ss11 * r01;
    float T11 = ss01 * r10 + ss11 * r11;
    // cov = R * T + I
    float a = r00 * T00 + r01 * T10 + 1.0f;
    float b = r00 * T01 + r01 * T11;
    float c = r10 * T00 + r11 * T10;
    float d = r10 * T01 + r11 * T11 + 1.0f;

    float det  = a * d - b * c;
    float rdet = 1.0f / det;

    ws[g * 8 + 0] = means[g * 2 + 0];                    // cx
    ws[g * 8 + 1] = means[g * 2 + 1];                    // cy
    ws[g * 8 + 2] = d * rdet;                            // inv00
    ws[g * 8 + 3] = -(b + c) * rdet;                     // inv01 + inv10
    ws[g * 8 + 4] = a * rdet;                            // inv11
    ws[g * 8 + 5] = 0.15915494309189535f * rsqrtf(det);  // 1/(2*pi*sqrt(det))
    ws[g * 8 + 6] = 0.0f;
    ws[g * 8 + 7] = 0.0f;
}

// ---------------------------------------------------------------------------
// Kernel 2: pack the (constant) WMMA B operand to f16 in exact lane layout.
// B is 32(K) x 16(N) per chunk; lane = column(0..15) + 16*khalf,
// element j holds K = j + 16*khalf. Columns >= 3 are zero. Branchless.
// One thread per (chunk, lane): 32 chunks * 32 lanes = 1024 threads.
// ---------------------------------------------------------------------------
__global__ __launch_bounds__(256) void gs2d_packB(
    const float* __restrict__ colors,    // (N,4)
    _Float16* __restrict__ bpk)          // 32*32*16 halves
{
    int t     = blockIdx.x * blockDim.x + threadIdx.x;   // 0..1023
    int chunk = t >> 5;
    int lane  = t & 31;
    int col   = lane & 15;
    int khalf = lane >> 4;

    int   cc   = (col < NCH) ? col : 0;                  // safe in-bounds column
    float mask = (col < NCH) ? 1.0f : 0.0f;

    v16h B;
    #pragma unroll
    for (int j = 0; j < 16; ++j) {
        int g = chunk * 32 + khalf * 16 + j;             // K -> gaussian id
        float alpha = colors[g * 4 + 3];
        float w = alpha / (1.0f + __expf(-colors[g * 4 + cc]));
        B[j] = (_Float16)(w * mask);
    }
    *(v16h*)&bpk[t * 16] = B;                            // 32B aligned store
}

// ---------------------------------------------------------------------------
// Kernel 3: render. Each wave32 owns 16 consecutive pixels (one WMMA M-tile).
// K=1024 gaussians in 32 chunks of 32, processed 2 chunks/iteration with
// two independent accumulators so consecutive WMMAs are not serialized on
// the accumulator (XDL latency hiding; f16 WMMA co-executes with VALU):
//   A (16 pix x 32 gauss, f16) = norm * exp(-0.5*quad), params from LDS (b128)
//   B loaded pre-packed from global (L2-resident, 32B/lane/chunk)
//   C += A x B via v_wmma_f32_16x16x32_f16
// ---------------------------------------------------------------------------
__global__ __launch_bounds__(256) void gs2d_render(
    const float* __restrict__ ws,
    float* __restrict__ out)             // (H, W, 3) flat
{
    __shared__ float sm[PARAM_FLOATS];   // 32 KB of the WGP's 320 KB LDS

    const int tid = threadIdx.x;
    // stage params global -> LDS as 16B vectors
    {
        const float4* src = (const float4*)ws;
        float4*       dst = (float4*)sm;
        for (int i = tid; i < PARAM_FLOATS / 4; i += 256) dst[i] = src[i];
    }
    __syncthreads();

    const int lane  = tid & 31;          // wave32
    const int wave  = tid >> 5;          // 8 waves per block
    const int base  = blockIdx.x * 128 + wave * 16;      // first pixel of tile

    const int m     = lane & 15;         // pixel row within tile
    const int khalf = lane >> 4;         // K-half select (contiguous k = 16*khalf + j)

    const int p = base + m;              // flat pixel id
    const float x = (float)(p >> 8) * (1.0f / 255.0f);   // xs[h], h = p / 256
    const float y = (float)(p & 255) * (1.0f / 255.0f);  // ys[w], w = p % 256

    const v16h* __restrict__ bptr = (const v16h*)(ws + PARAM_FLOATS);

    v8f acc0 = {};
    v8f acc1 = {};

    for (int g0 = 0; g0 < NG; g0 += 64) {
        // ---- B operands for both chunks: 32B vector loads, issued early ----
        v16h B0 = bptr[((g0      ) >> 5) * 32 + lane];
        v16h B1 = bptr[((g0 + 32 ) >> 5) * 32 + lane];

        // ---- A matrices: element j holds K = j + 16*khalf ----
        v16h A0, A1;
        #pragma unroll
        for (int j = 0; j < 16; ++j) {
            int k = j + khalf * 16;

            {
                int g = g0 + k;
                float4 p0 = *(const float4*)&sm[g * 8];      // cx, cy, i00, i01s
                float4 p1 = *(const float4*)&sm[g * 8 + 4];  // i11, norm, -, -
                float dx = x - p0.x;
                float dy = y - p0.y;
                float q  = p0.z * dx * dx + p0.w * dx * dy + p1.x * dy * dy;
                A0[j] = (_Float16)(p1.y * __expf(-0.5f * q));
            }
            {
                int g = g0 + 32 + k;
                float4 p0 = *(const float4*)&sm[g * 8];
                float4 p1 = *(const float4*)&sm[g * 8 + 4];
                float dx = x - p0.x;
                float dy = y - p0.y;
                float q  = p0.z * dx * dx + p0.w * dx * dy + p1.x * dy * dy;
                A1[j] = (_Float16)(p1.y * __expf(-0.5f * q));
            }
        }

        // Two independent accumulator chains (EXEC all-1s on this path)
        acc0 = __builtin_amdgcn_wmma_f32_16x16x32_f16(
            false, A0, false, B0, (short)0, acc0, false, false);
        acc1 = __builtin_amdgcn_wmma_f32_16x16x32_f16(
            false, A1, false, B1, (short)0, acc1, false, false);
    }

    v8f acc = acc0 + acc1;

    // ---- scatter D: VGPR r -> row r (lanes 0-15) / row r+8 (lanes 16-31),
    //      column = lane / lane-16; only columns 0..2 are live ----
    int ch = -1, rowoff = 0;
    if (lane < NCH)                        { ch = lane;      rowoff = 0; }
    else if (lane >= 16 && lane < 16+NCH)  { ch = lane - 16; rowoff = 8; }
    if (ch >= 0) {
        #pragma unroll
        for (int r = 0; r < 8; ++r) {
            int pix = base + rowoff + r;
            out[pix * NCH + ch] = acc[r];
        }
    }
}

// ---------------------------------------------------------------------------
extern "C" void kernel_launch(void* const* d_in, const int* in_sizes, int n_in,
                              void* d_out, int out_size, void* d_ws, size_t ws_size,
                              hipStream_t stream) {
    const float* scale  = (const float*)d_in[0];
    const float* rot    = (const float*)d_in[1];
    const float* means  = (const float*)d_in[2];
    const float* colors = (const float*)d_in[3];
    float* ws  = (float*)d_ws;
    float* out = (float*)d_out;

    gs2d_prep <<<NG / 256, 256, 0, stream>>>(scale, rot, means, ws);
    gs2d_packB<<<NG / 256, 256, 0, stream>>>(colors, (_Float16*)(ws + PARAM_FLOATS));

    // 65536 pixels / (8 waves * 16 pixels) = 512 blocks
    gs2d_render<<<(IMG_H * IMG_W) / 128, 256, 0, stream>>>(ws, out);

    (void)in_sizes; (void)n_in; (void)out_size; (void)ws_size;
}
